// attention_model_45088566673900
// MI455X (gfx1250) — compile-verified, matrix operand order checked by reading, not run on hardware
//
#include <hip/hip_runtime.h>
#include <hip/hip_bf16.h>
#include <math.h>

// ---------------------------------------------------------------------------
// LSTM encoder/decoder with additive attention, greedy decode — MI455X.
//  * bf16 weight cache in workspace: out_W(65.5MB)+dec(25MB)+enc(17MB)+misc
//    all fit the 192MB L2 -> sequential decode streams from L2, not HBM.
//  * WMMA (v_wmma_f32_16x16x32_bf16) for BOTH real GEMMs:
//      Gx  = x_emb @ enc_Wih^T   (256 x 4096 x 1024)  <- hoisted out of the
//            encoder time loop (input projection is batchable)
//      Ew2 = E @ w2^T            (256 x 1024 x 1024)
//  * Sequential parts are wave-per-row bf16 dot products (wave32 shfl reduce).
//  * Aggressive kernel fusion to shrink the graph: 1 kernel/encoder step,
//    6 kernels/decoder step (~1.8k nodes total).
// ---------------------------------------------------------------------------

typedef __attribute__((ext_vector_type(16))) __bf16 v16bf;
typedef __attribute__((ext_vector_type(4)))  __bf16 v4bf;
typedef __attribute__((ext_vector_type(8)))  float  v8f;

#define HDIM 1024
#define G4H  4096      // 4*H
#define LSEQ 256
#define VOC  32000

__device__ __forceinline__ float waveSum(float v) {
#pragma unroll
  for (int off = 16; off > 0; off >>= 1) v += __shfl_down(v, off, 32);
  return v;
}
__device__ __forceinline__ float sigmoidf_(float x) { return 1.0f / (1.0f + expf(-x)); }

// ---------------- elementwise utilities ----------------
__global__ void k_f32_to_bf16(const float* __restrict__ src, __bf16* __restrict__ dst, int n) {
  int i = blockIdx.x * blockDim.x + threadIdx.x;
  if (i < n) dst[i] = (__bf16)src[i];
}

__global__ void k_init_state(float* __restrict__ hA, float* __restrict__ c,
                             float* __restrict__ prev) {
  int j = blockIdx.x * blockDim.x + threadIdx.x;
  if (j < HDIM) { hA[j] = 0.0f; c[j] = 0.0f; prev[j] = 0.0f; }
}

// embedding gather, directly to bf16 (feeds the Gx WMMA GEMM)
__global__ void k_gather_embeds_bf16(const int* __restrict__ seq,
                                     const float* __restrict__ embeds,
                                     __bf16* __restrict__ x_emb_bf) {
  int i = blockIdx.x * blockDim.x + threadIdx.x;   // L*H total
  int l = i >> 10, j = i & (HDIM - 1);
  if (i < LSEQ * HDIM) x_emb_bf[i] = (__bf16)embeds[(size_t)seq[l] * HDIM + j];
}

// ---------------- generic WMMA GEMM:  C[M,N] = A[M,K] @ B[N,K]^T ----------------
// One wave per 16x16 output tile; v_wmma_f32_16x16x32_bf16 K-loop.
// Per-lane element coordinates follow CDNA5 ISA §7.12.2 (the gathers are two
// contiguous 16B chunks per operand -> compiler emits global_load_b128 pairs).
__global__ void k_gemm_bf16_nt(const __bf16* __restrict__ A,   // (M,K) row-major
                               const __bf16* __restrict__ B,   // (N,K) row-major
                               float* __restrict__ C,          // (M,N) row-major
                               int N, int K) {
  int tilesN = N >> 4;
  int wave = (blockIdx.x * blockDim.x + threadIdx.x) >> 5;   // grid sized exactly
  int lane = threadIdx.x & 31;
  int tm = wave / tilesN;
  int tn = wave % tilesN;
  int half = lane >> 4;                               // 0/1
  int mn   = lane & 15;                               // row (A) / col (B,D)
  v8f acc = {};
  for (int k0 = 0; k0 < K; k0 += 32) {
    v16bf a, b;
#pragma unroll
    for (int i = 0; i < 16; ++i) {
      int v = i >> 1, pos = i & 1;
      int k = (v < 4) ? (half * 8 + 2 * v + pos)
                      : (16 + half * 8 + 2 * (v - 4) + pos);
      a[i] = A[(size_t)(tm * 16 + mn) * K + (k0 + k)];
      b[i] = B[(size_t)(tn * 16 + mn) * K + (k0 + k)];
    }
    acc = __builtin_amdgcn_wmma_f32_16x16x32_bf16(
        /*neg_a=*/false, a, /*neg_b=*/false, b,
        /*c_mod=*/(short)0, acc, /*reuse_a=*/false, /*reuse_b=*/false);
  }
#pragma unroll
  for (int r = 0; r < 8; ++r) {
    int m = r + half * 8;
    C[(size_t)(tm * 16 + m) * N + tn * 16 + mn] = acc[r];
  }
}

// ---------------- fused LSTM cell (matvec + gates in one kernel) ----------------
// Block = 256 threads = 8 waves = {2 columns} x {4 gates i,f,g,o}.
// g[row] = (Wih? Wih[row,:]@x : 0) + Whh[row,:]@hin + (Gpre? Gpre[row] : 0)
//          + bih[row] + bhh[row],   row = gate*H + j
// Then threads 0..1 apply the gate nonlinearity for their column j.
// hin is read-only, hout write-only (double buffered across steps: no race);
// c[j] is touched only by the block owning j -> safe in place.
__global__ void k_lstm_cell(const __bf16* __restrict__ Wih, int Kx,
                            const __bf16* __restrict__ Whh,
                            const float* __restrict__ x,
                            const float* __restrict__ hin,
                            const float* __restrict__ Gpre,
                            const float* __restrict__ bih, const float* __restrict__ bhh,
                            float* __restrict__ hout, float* __restrict__ c,
                            float* __restrict__ Erow) {
  __shared__ float g8[8];
  int w    = threadIdx.x >> 5;          // wave 0..7
  int lane = threadIdx.x & 31;
  int gate = w & 3;
  int jj   = w >> 2;
  int j    = blockIdx.x * 2 + jj;
  int row  = gate * HDIM + j;
  float s = 0.0f;
  if (Wih) {
    const __bf16* r = Wih + (size_t)row * Kx;
    for (int k = lane * 4; k < Kx; k += 128) {
      v4bf  wv = *(const v4bf*)(r + k);
      float4 xv = *(const float4*)(x + k);
      s = fmaf((float)wv[0], xv.x, s); s = fmaf((float)wv[1], xv.y, s);
      s = fmaf((float)wv[2], xv.z, s); s = fmaf((float)wv[3], xv.w, s);
    }
  }
  {
    const __bf16* r = Whh + (size_t)row * HDIM;
    for (int k = lane * 4; k < HDIM; k += 128) {
      v4bf  wv = *(const v4bf*)(r + k);
      float4 hv = *(const float4*)(hin + k);
      s = fmaf((float)wv[0], hv.x, s); s = fmaf((float)wv[1], hv.y, s);
      s = fmaf((float)wv[2], hv.z, s); s = fmaf((float)wv[3], hv.w, s);
    }
  }
  s = waveSum(s);
  if (lane == 0)
    g8[w] = s + bih[row] + bhh[row] + (Gpre ? Gpre[row] : 0.0f);
  __syncthreads();
  int t = threadIdx.x;
  if (t < 2) {
    int jc = blockIdx.x * 2 + t;
    float ig = sigmoidf_(g8[t * 4 + 0]);
    float fg = sigmoidf_(g8[t * 4 + 1]);
    float gg = tanhf   (g8[t * 4 + 2]);
    float og = sigmoidf_(g8[t * 4 + 3]);
    float cn = fg * c[jc] + ig * gg;
    float hn = og * tanhf(cn);
    c[jc] = cn; hout[jc] = hn;
    if (Erow) Erow[jc] = hn;
  }
}

// ---------------- wave-per-row matvec (w1 @ h) ----------------
__global__ void k_matvec_bf16(const __bf16* __restrict__ W, const float* __restrict__ x,
                              float* __restrict__ y, int rows, int K) {
  int wave = (blockIdx.x * blockDim.x + threadIdx.x) >> 5;
  int lane = threadIdx.x & 31;
  if (wave >= rows) return;
  const __bf16* row = W + (size_t)wave * K;
  float s = 0.0f;
  for (int k = lane * 4; k < K; k += 128) {
    v4bf  w  = *(const v4bf*)(row + k);
    float4 xv = *(const float4*)(x + k);
    s = fmaf((float)w[0], xv.x, s); s = fmaf((float)w[1], xv.y, s);
    s = fmaf((float)w[2], xv.z, s); s = fmaf((float)w[3], xv.w, s);
  }
  s = waveSum(s);
  if (lane == 0) y[wave] = s;
}

// ---------------- attention scores: one wave per l ----------------
__global__ void k_attn_scores(const float* __restrict__ w1h, const float* __restrict__ Ew2,
                              const float* __restrict__ v, float* __restrict__ scores) {
  int wave = (blockIdx.x * blockDim.x + threadIdx.x) >> 5;
  int lane = threadIdx.x & 31;
  if (wave >= LSEQ) return;
  const float* row = Ew2 + (size_t)wave * HDIM;
  float s = 0.0f;
  for (int k = lane; k < HDIM; k += 32)
    s = fmaf(tanhf(w1h[k] + row[k]), v[k], s);
  s = waveSum(s);
  if (lane == 0) scores[wave] = s;
}

// ---------------- fused softmax + context + xin build (1 block, 1024 thr) ----------
__global__ void k_softmax_ctx(const float* __restrict__ scores, const float* __restrict__ E,
                              const float* __restrict__ prev, float* __restrict__ xin) {
  __shared__ float sa[LSEQ];
  __shared__ float tmp[LSEQ];
  int t = threadIdx.x;                 // 0..1023 (H exactly)
  if (t < LSEQ) { sa[t] = scores[t]; tmp[t] = sa[t]; }
  __syncthreads();
  for (int str = 128; str > 0; str >>= 1) {
    if (t < str) tmp[t] = fmaxf(tmp[t], tmp[t + str]);
    __syncthreads();
  }
  float mx = tmp[0];
  __syncthreads();
  if (t < LSEQ) { sa[t] = expf(sa[t] - mx); tmp[t] = sa[t]; }
  __syncthreads();
  for (int str = 128; str > 0; str >>= 1) {
    if (t < str) tmp[t] += tmp[t + str];
    __syncthreads();
  }
  float inv = 1.0f / tmp[0];
  float s = 0.0f;
  for (int l = 0; l < LSEQ; ++l) s = fmaf(sa[l], E[(size_t)l * HDIM + t], s);
  xin[HDIM + t] = s * inv;             // ctx
  xin[t] = fmaxf(prev[t], 0.0f);       // relu(prev embedding)
}

// ---------------- logits (wave-per-row) + in-block argmax partial ----------------
__global__ void k_logits_partial(const __bf16* __restrict__ W, const float* __restrict__ h2,
                                 const float* __restrict__ bias, float* __restrict__ out,
                                 float* __restrict__ pv, int* __restrict__ pi) {
  __shared__ float sv[8];
  __shared__ int   si[8];
  int w    = threadIdx.x >> 5;                    // 8 waves
  int lane = threadIdx.x & 31;
  int r    = blockIdx.x * 8 + w;                  // grid covers VOC exactly
  const __bf16* row = W + (size_t)r * HDIM;
  float s = 0.0f;
  for (int k = lane * 4; k < HDIM; k += 128) {
    __builtin_prefetch(row + k + 512, 0, 0);      // -> global_prefetch_b8
    v4bf  wv = *(const v4bf*)(row + k);
    float4 hv = *(const float4*)(h2 + k);
    s = fmaf((float)wv[0], hv.x, s); s = fmaf((float)wv[1], hv.y, s);
    s = fmaf((float)wv[2], hv.z, s); s = fmaf((float)wv[3], hv.w, s);
  }
  s = waveSum(s);
  if (lane == 0) {
    float val = s + bias[r];
    out[r] = val;
    sv[w] = val; si[w] = r;
  }
  __syncthreads();
  if (threadIdx.x == 0) {
    float bv = sv[0]; int bi = si[0];
    for (int q = 1; q < 8; ++q)
      if (sv[q] > bv) { bv = sv[q]; bi = si[q]; }   // ascending -> first max wins
    pv[blockIdx.x] = bv; pi[blockIdx.x] = bi;
  }
}

// ---------------- final argmax over partials + greedy feedback ----------------
__global__ void k_argmax_final(const float* __restrict__ pv, const int* __restrict__ pi,
                               int nparts, const float* __restrict__ embeds,
                               float* __restrict__ prev, float* __restrict__ out_id) {
  __shared__ float sv[1024];
  __shared__ int   si[1024];
  __shared__ int   best;
  int t = threadIdx.x;
  float v = -__builtin_inff(); int ix = 0x7fffffff;
  for (int p = t; p < nparts; p += 1024) {
    float v2 = pv[p]; int i2 = pi[p];
    if (v2 > v || (v2 == v && i2 < ix)) { v = v2; ix = i2; }
  }
  sv[t] = v; si[t] = ix;
  __syncthreads();
  for (int str = 512; str > 0; str >>= 1) {
    if (t < str) {
      float v2 = sv[t + str]; int i2 = si[t + str];
      if (v2 > sv[t] || (v2 == sv[t] && i2 < si[t])) { sv[t] = v2; si[t] = i2; }
    }
    __syncthreads();
  }
  if (t == 0) { best = si[0]; *out_id = (float)si[0]; }
  __syncthreads();
  prev[t] = embeds[(size_t)best * HDIM + t];      // blockDim == H
}

// ---------------------------------------------------------------------------
extern "C" void kernel_launch(void* const* d_in, const int* in_sizes, int n_in,
                              void* d_out, int out_size, void* d_ws, size_t ws_size,
                              hipStream_t stream) {
  (void)in_sizes; (void)n_in; (void)out_size; (void)ws_size;
  const int*   input_seq = (const int*)  d_in[0];
  const float* embeds    = (const float*)d_in[1];
  const float* enc_Wih   = (const float*)d_in[2];
  const float* enc_Whh   = (const float*)d_in[3];
  const float* enc_bih   = (const float*)d_in[4];
  const float* enc_bhh   = (const float*)d_in[5];
  const float* dec_Wih   = (const float*)d_in[6];
  const float* dec_Whh   = (const float*)d_in[7];
  const float* dec_bih   = (const float*)d_in[8];
  const float* dec_bhh   = (const float*)d_in[9];
  const float* out_W     = (const float*)d_in[10];
  const float* out_b     = (const float*)d_in[11];
  const float* vvec      = (const float*)d_in[12];
  const float* w1        = (const float*)d_in[13];
  const float* w2        = (const float*)d_in[14];
  float* out = (float*)d_out;                 // [L*V logits] then [L ids-as-float]

  // ---- workspace carve (~120 MB; bf16 cache keeps working set L2-resident)
  char* ws = (char*)d_ws; size_t off = 0;
  auto carve = [&](size_t bytes) -> char* {
    char* p = ws + off; off += (bytes + 255) & ~(size_t)255; return p;
  };
  __bf16* enc_Wih_bf = (__bf16*)carve((size_t)G4H * HDIM * 2);
  __bf16* enc_Whh_bf = (__bf16*)carve((size_t)G4H * HDIM * 2);
  __bf16* dec_Wih_bf = (__bf16*)carve((size_t)G4H * 2 * HDIM * 2);
  __bf16* dec_Whh_bf = (__bf16*)carve((size_t)G4H * HDIM * 2);
  __bf16* outW_bf    = (__bf16*)carve((size_t)VOC * HDIM * 2);
  __bf16* w1_bf      = (__bf16*)carve((size_t)HDIM * HDIM * 2);
  __bf16* w2_bf      = (__bf16*)carve((size_t)HDIM * HDIM * 2);
  __bf16* xemb_bf    = (__bf16*)carve((size_t)LSEQ * HDIM * 2);
  __bf16* E_bf       = (__bf16*)carve((size_t)LSEQ * HDIM * 2);
  float* Gx     = (float*)carve((size_t)LSEQ * G4H * 4);   // x_emb @ enc_Wih^T
  float* Ebuf   = (float*)carve((size_t)LSEQ * HDIM * 4);
  float* Ew2    = (float*)carve((size_t)LSEQ * HDIM * 4);
  float* hA     = (float*)carve(HDIM * 4);
  float* hB     = (float*)carve(HDIM * 4);
  float* cbuf   = (float*)carve(HDIM * 4);
  float* prev   = (float*)carve(HDIM * 4);
  float* w1h    = (float*)carve(HDIM * 4);
  float* scores = (float*)carve(LSEQ * 4);
  float* xin    = (float*)carve(2 * HDIM * 4);
  float* pval   = (float*)carve(4096 * 4);
  int*   pidx   = (int*)  carve(4096 * 4);

  const int TB = 256;
  auto cvt = [&](const float* s, __bf16* d, int n) {
    k_f32_to_bf16<<<(n + TB - 1) / TB, TB, 0, stream>>>(s, d, n);
  };

  // ---- one-time (per call) bf16 weight cache
  cvt(enc_Wih, enc_Wih_bf, G4H * HDIM);
  cvt(enc_Whh, enc_Whh_bf, G4H * HDIM);
  cvt(dec_Wih, dec_Wih_bf, G4H * 2 * HDIM);
  cvt(dec_Whh, dec_Whh_bf, G4H * HDIM);
  cvt(out_W,   outW_bf,    VOC * HDIM);
  cvt(w1,      w1_bf,      HDIM * HDIM);
  cvt(w2,      w2_bf,      HDIM * HDIM);

  k_gather_embeds_bf16<<<(LSEQ * HDIM) / TB, TB, 0, stream>>>(input_seq, embeds, xemb_bf);
  k_init_state<<<HDIM / TB, TB, 0, stream>>>(hA, cbuf, prev);

  // ---- WMMA GEMM 1: Gx = x_emb @ enc_Wih^T  (hoisted encoder input projection)
  {
    int tiles = (LSEQ / 16) * (G4H / 16);        // 4096 waves
    k_gemm_bf16_nt<<<(tiles * 32) / 128, 128, 0, stream>>>(xemb_bf, enc_Wih_bf, Gx, G4H, HDIM);
  }

  // ---- encoder LSTM: one fused kernel per step (Whh matvec + gates)
  for (int t = 0; t < LSEQ; ++t) {
    float* hin  = (t & 1) ? hB : hA;
    float* hout = (t & 1) ? hA : hB;
    k_lstm_cell<<<HDIM / 2, TB, 0, stream>>>(
        nullptr, 0, enc_Whh_bf, nullptr, hin, Gx + (size_t)t * G4H,
        enc_bih, enc_bhh, hout, cbuf, Ebuf + (size_t)t * HDIM);
  }
  // final encoder h is in hA (t=255 wrote hA); decoder continues same parity.

  // ---- WMMA GEMM 2: Ew2 = E @ w2^T
  cvt(Ebuf, E_bf, LSEQ * HDIM);
  {
    int tiles = (LSEQ / 16) * (HDIM / 16);       // 1024 waves
    k_gemm_bf16_nt<<<(tiles * 32) / 128, 128, 0, stream>>>(E_bf, w2_bf, Ew2, HDIM, HDIM);
  }

  // ---- greedy autoregressive decoder: 6 fused kernels per step
  for (int t = 0; t < LSEQ; ++t) {
    float* hin  = (t & 1) ? hB : hA;
    float* hout = (t & 1) ? hA : hB;
    k_matvec_bf16<<<(HDIM * 32) / TB, TB, 0, stream>>>(w1_bf, hin, w1h, HDIM, HDIM);
    k_attn_scores<<<(LSEQ * 32) / TB, TB, 0, stream>>>(w1h, Ew2, vvec, scores);
    k_softmax_ctx<<<1, 1024, 0, stream>>>(scores, Ebuf, prev, xin);
    k_lstm_cell<<<HDIM / 2, TB, 0, stream>>>(
        dec_Wih_bf, 2 * HDIM, dec_Whh_bf, xin, hin, nullptr,
        dec_bih, dec_bhh, hout, cbuf, nullptr);
    k_logits_partial<<<VOC / 8, TB, 0, stream>>>(
        outW_bf, hout, out_b, out + (size_t)t * VOC, pval, pidx);
    k_argmax_final<<<1, 1024, 0, stream>>>(
        pval, pidx, VOC / 8, embeds, prev, out + (size_t)LSEQ * VOC + t);
  }
}
